// FastHelgasonMLP_70729521431228
// MI455X (gfx1250) — matrix operand order repaired and checked by reading, not compiled, not run on hardware
//
#include <hip/hip_runtime.h>
#include <hip/hip_bf16.h>

// Factorized low-rank linear: out = x @ V * S @ U^T   (fp32 in/out, bf16 WMMA core)
// ws layout (48 MB needed): Vt bf16 [1024][4096] | Ub bf16 [4096][1024] | Yp bf16 [16384][1024]

typedef __bf16 bf16_t;
typedef bf16_t v16bf __attribute__((ext_vector_type(16)));
typedef bf16_t v8bf  __attribute__((ext_vector_type(8)));
typedef bf16_t v4bf  __attribute__((ext_vector_type(4)));
typedef float  v8f   __attribute__((ext_vector_type(8)));
typedef float  v4f   __attribute__((ext_vector_type(4)));

#define OUT_DIM 4096
#define IN_DIM  4096
#define RANK    1024
#define BT      16384  // B*T = 4*4096

// --- CDNA5 async global->LDS helpers (ASYNCcnt-tracked, bypasses VGPRs) ---

__device__ __forceinline__ unsigned lds_off(const void* p) {
  // generic -> addrspace(3) cast yields the raw 32-bit LDS byte offset
  return (unsigned)(unsigned long long)(__attribute__((address_space(3))) const void*)p;
}

__device__ __forceinline__ void async_copy_b128(const void* g, const void* l) {
  unsigned lofs = lds_off(l);
  unsigned long long ga = (unsigned long long)g;
  asm volatile("global_load_async_to_lds_b128 %0, %1, off"
               :: "v"(lofs), "v"(ga) : "memory");
}

__device__ __forceinline__ void wait_async0() {
  asm volatile("s_wait_asynccnt 0x0" ::: "memory");
}

// ---------------- prep kernels ----------------

__global__ void convert_u_kernel(const float* __restrict__ U, bf16_t* __restrict__ Ub, int n) {
  int i = blockIdx.x * blockDim.x + threadIdx.x;
  int stride = gridDim.x * blockDim.x;
  for (; i < n; i += stride) Ub[i] = (bf16_t)U[i];
}

// V [IN][RANK] fp32 -> Vt [RANK][IN] bf16, LDS-tiled 32x32 transpose
__global__ void transpose_v_kernel(const float* __restrict__ V, bf16_t* __restrict__ Vt) {
  __shared__ float tile[32][33];
  const int tx = threadIdx.x, ty = threadIdx.y;
  const int r0 = blockIdx.x * 32;   // rank dim
  const int i0 = blockIdx.y * 32;   // in dim
#pragma unroll
  for (int j = 0; j < 32; j += 8)
    tile[ty + j][tx] = V[(size_t)(i0 + ty + j) * RANK + r0 + tx];
  __syncthreads();
#pragma unroll
  for (int j = 0; j < 32; j += 8)
    Vt[(size_t)(r0 + ty + j) * IN_DIM + i0 + tx] = (bf16_t)tile[tx][ty + j];
}

// ---------------- WMMA GEMM ----------------
// C[M,N] = A[M,K] @ B[N,K]^T ; B stored row-major [N][K] (each row = one B^T row).
// STAGE1: A = fp32 x (convert on the fly via VGPR staging), epilogue scales col by S, stores bf16.
// STAGE2: A = bf16 Yp (async-to-LDS), stores fp32.
// B tiles always bf16 -> async-to-LDS. One barrier per k-step.
// Block = 256 threads (8 waves), tile 128x128, K-step 32, double-buffered LDS.

#define LDS_STRIDE 40  // 32 + 8 halves pad: 80B rows -> conflict-free b128 lane pattern

template <bool STAGE1>
__global__ __launch_bounds__(256) void gemm_wmma(
    const float* __restrict__ Af32, const bf16_t* __restrict__ Abf,
    const bf16_t* __restrict__ Bmat, const float* __restrict__ S,
    bf16_t* __restrict__ Obf, float* __restrict__ Of32, int K, int N) {
  __shared__ __align__(16) bf16_t sA[2][128][LDS_STRIDE];
  __shared__ __align__(16) bf16_t sB[2][128][LDS_STRIDE];

  const int tid   = threadIdx.x;
  const int lane  = tid & 31;
  const int wave  = tid >> 5;   // 0..7
  const int waveM = wave >> 2;  // 0..1 -> 64 rows each
  const int waveN = wave & 3;   // 0..3 -> 32 cols each

  const int rowBase = blockIdx.y * 128;
  const int colBase = blockIdx.x * 128;

  v8f acc[4][2];
#pragma unroll
  for (int f = 0; f < 4; ++f)
#pragma unroll
    for (int g = 0; g < 2; ++g)
#pragma unroll
      for (int j = 0; j < 8; ++j) acc[f][g][j] = 0.0f;

  // STAGE1 A staging registers (fp32 -> bf16 conversion must go through VALU)
  v4f aRegF[4];

  auto issueB = [&](int kBase, int buf) {
#pragma unroll
    for (int i = 0; i < 2; ++i) {
      int s = tid + 256 * i;           // 512 b128 slots (128 rows x 4 chunks)
      int r = s >> 2, kq = s & 3;
      async_copy_b128(Bmat + (size_t)(colBase + r) * K + kBase + kq * 8,
                      &sB[buf][r][kq * 8]);
    }
  };
  auto issueA_bf16 = [&](int kBase, int buf) {
#pragma unroll
    for (int i = 0; i < 2; ++i) {
      int s = tid + 256 * i;
      int r = s >> 2, kq = s & 3;
      async_copy_b128(Abf + (size_t)(rowBase + r) * K + kBase + kq * 8,
                      &sA[buf][r][kq * 8]);
    }
  };
  auto fetchA_f32 = [&](int kBase) {
#pragma unroll
    for (int i = 0; i < 4; ++i) {
      int s = tid + 256 * i;           // 1024 float4 slots
      int r = s >> 3, kq = s & 7;      // 8 float4 per 32-wide row
      aRegF[i] = *(const v4f*)(Af32 + (size_t)(rowBase + r) * K + kBase + kq * 4);
    }
  };
  auto stageA_f32 = [&](int buf) {
#pragma unroll
    for (int i = 0; i < 4; ++i) {
      int s = tid + 256 * i;
      int r = s >> 3, kq = s & 7;
      v4f fv = aRegF[i];
      v4bf hv;
      hv[0] = (bf16_t)fv[0]; hv[1] = (bf16_t)fv[1];
      hv[2] = (bf16_t)fv[2]; hv[3] = (bf16_t)fv[3];
      *(v4bf*)&sA[buf][r][kq * 4] = hv;
    }
  };

  union AV { v16bf v; struct { v8bf lo, hi; } p; };

  const int mrow  = lane & 15;
  const int aKoff = (lane >> 4) * 8;   // ISA A layout: lanes<16 K{0-7,16-23}, lanes>=16 K{8-15,24-31}
  const int bKoff = (lane >> 4) * 8;   // B layout matches A: lanes<16 K{0-7,16-23}, lanes>=16 K{8-15,24-31} of column N=lane&15

  auto compute = [&](int buf) {
    AV a[4];
#pragma unroll
    for (int f = 0; f < 4; ++f) {
      const bf16_t* base = &sA[buf][waveM * 64 + f * 16 + mrow][0];
      a[f].p.lo = *(const v8bf*)(base + aKoff);
      a[f].p.hi = *(const v8bf*)(base + aKoff + 16);
    }
    AV b[2];
#pragma unroll
    for (int g = 0; g < 2; ++g) {
      const bf16_t* base = &sB[buf][waveN * 32 + g * 16 + mrow][0];
      b[g].p.lo = *(const v8bf*)(base + bKoff);
      b[g].p.hi = *(const v8bf*)(base + bKoff + 16);
    }
#pragma unroll
    for (int f = 0; f < 4; ++f)
#pragma unroll
      for (int g = 0; g < 2; ++g)
        acc[f][g] = __builtin_amdgcn_wmma_f32_16x16x32_bf16(
            false, a[f].v, false, b[g].v, (short)0, acc[f][g], false, false);
  };

  const int nK = K / 32;

  // Prologue: fill buffer 0
  issueB(0, 0);
  if (STAGE1) { fetchA_f32(0); stageA_f32(0); }
  else        { issueA_bf16(0, 0); }
  wait_async0();
  __syncthreads();

#pragma unroll 1
  for (int kt = 0; kt < nK; ++kt) {
    const int cur = kt & 1;
    if (kt + 1 < nK) {
      issueB((kt + 1) * 32, cur ^ 1);              // async, straight into LDS
      if (STAGE1) fetchA_f32((kt + 1) * 32);       // regs; overlapped with WMMAs
      else        issueA_bf16((kt + 1) * 32, cur ^ 1);
    }
    compute(cur);
    if (STAGE1 && (kt + 1 < nK)) stageA_f32(cur ^ 1);  // cvt + LDS store after compute
    wait_async0();
    __syncthreads();
  }

  // Epilogue. C/D layout: VGPR j -> M = j + 8*(lane>=16), N = lane&15.
  const int col0 = colBase + waveN * 32 + (lane & 15);
  const int rsub = (lane >> 4) * 8;
  if (STAGE1) {
    const float s0 = S[col0];
    const float s1 = S[col0 + 16];
#pragma unroll
    for (int f = 0; f < 4; ++f) {
      const int row = rowBase + waveM * 64 + f * 16 + rsub;
#pragma unroll
      for (int j = 0; j < 8; ++j) {
        Obf[(size_t)(row + j) * N + col0]      = (bf16_t)(acc[f][0][j] * s0);
        Obf[(size_t)(row + j) * N + col0 + 16] = (bf16_t)(acc[f][1][j] * s1);
      }
    }
  } else {
#pragma unroll
    for (int f = 0; f < 4; ++f) {
      const int row = rowBase + waveM * 64 + f * 16 + rsub;
#pragma unroll
      for (int j = 0; j < 8; ++j) {
        Of32[(size_t)(row + j) * N + col0]      = acc[f][0][j];
        Of32[(size_t)(row + j) * N + col0 + 16] = acc[f][1][j];
      }
    }
  }
}

// ---------------- launch ----------------

extern "C" void kernel_launch(void* const* d_in, const int* in_sizes, int n_in,
                              void* d_out, int out_size, void* d_ws, size_t ws_size,
                              hipStream_t stream) {
  const float* x = (const float*)d_in[0];  // [BT][IN]
  const float* U = (const float*)d_in[1];  // [OUT][RANK]
  const float* S = (const float*)d_in[2];  // [RANK]
  const float* V = (const float*)d_in[3];  // [IN][RANK]

  bf16_t* Vt = (bf16_t*)d_ws;                       //  8 MB  [RANK][IN]
  bf16_t* Ub = Vt + (size_t)RANK * IN_DIM;          //  8 MB  [OUT][RANK]
  bf16_t* Yp = Ub + (size_t)OUT_DIM * RANK;         // 32 MB  [BT][RANK]

  convert_u_kernel<<<2048, 256, 0, stream>>>(U, Ub, OUT_DIM * RANK);
  transpose_v_kernel<<<dim3(RANK / 32, IN_DIM / 32), dim3(32, 8), 0, stream>>>(V, Vt);

  // Stage 1: Yp[m,r] = S[r] * sum_i x[m,i] * Vt[r,i]
  gemm_wmma<true><<<dim3(RANK / 128, BT / 128), 256, 0, stream>>>(
      x, nullptr, Vt, S, Yp, nullptr, IN_DIM, RANK);

  // Stage 2: out[m,o] = sum_r Yp[m,r] * Ub[o,r]
  gemm_wmma<false><<<dim3(OUT_DIM / 128, BT / 128), 256, 0, stream>>>(
      nullptr, Yp, Ub, nullptr, nullptr, (float*)d_out, RANK, OUT_DIM);
}
